// Multi_Head_Attention_81484119539794
// MI455X (gfx1250) — matrix operand order repaired and checked by reading, not compiled, not run on hardware
//
#include <hip/hip_runtime.h>

// ---------------------------------------------------------------------------
// MI455X (gfx1250, wave32) multi-head attention + RPE + projection.
//   - big contractions on v_wmma_f32_16x16x32_f16 (f16 in, f32 accumulate)
//   - RPE einsum collapsed to a 129-entry per-row precompute + gather
//   - flash-style streaming softmax (no n x n score tensor in HBM)
//   - tile staging via GLOBAL_LOAD_ASYNC_TO_LDS_B128 (ASYNCcnt)
//   - global_prefetch of the next K/V tile (L2-resident working set)
// ---------------------------------------------------------------------------

#define B_  4
#define N_  2048
#define E_  512
#define H_  8
#define D_  64
#define BH_ (B_*H_)
#define RPE_N 129
#define RPE_STRIDE 132

typedef _Float16 h16;
typedef _Float16 v8h  __attribute__((ext_vector_type(8)));
typedef _Float16 v16h __attribute__((ext_vector_type(16)));
typedef float    v8f  __attribute__((ext_vector_type(8)));
typedef int      b128_t __attribute__((vector_size(16)));

static __device__ __forceinline__ v16h join16(v8h lo, v8h hi) {
  return __builtin_shufflevector(lo, hi, 0,1,2,3,4,5,6,7,8,9,10,11,12,13,14,15);
}

// ---- async global->LDS staging (CDNA5 ASYNCcnt path), guarded fallback ----
#if __has_builtin(__builtin_amdgcn_global_load_async_to_lds_b128)
static __device__ __forceinline__ void async_copy16(void* lds_dst, const void* gsrc) {
  __builtin_amdgcn_global_load_async_to_lds_b128(
      (__attribute__((address_space(1))) b128_t*)gsrc,
      (__attribute__((address_space(3))) b128_t*)lds_dst, 0, 0);
}
static __device__ __forceinline__ void async_wait() {
#if __has_builtin(__builtin_amdgcn_s_wait_asynccnt)
  __builtin_amdgcn_s_wait_asynccnt(0);
#else
  asm volatile("s_wait_asynccnt 0x0" ::: "memory");
#endif
}
#else
static __device__ __forceinline__ void async_copy16(void* lds_dst, const void* gsrc) {
  *(uint4*)lds_dst = *(const uint4*)gsrc;
}
static __device__ __forceinline__ void async_wait() {}
#endif

// --------------------------- fp32 -> f16 convert ---------------------------
__global__ void f32_to_f16_kernel(const float* __restrict__ src,
                                  h16* __restrict__ dst, int count) {
  int i = blockIdx.x * blockDim.x + threadIdx.x;
  if (i < count) dst[i] = (h16)src[i];
}

// ------------------------- LayerNorm(V) -> f16 -----------------------------
__global__ __launch_bounds__(256) void layernorm_v_kernel(
    const float* __restrict__ v, const float* __restrict__ gamma,
    const float* __restrict__ beta, h16* __restrict__ out) {
  int wave = threadIdx.x >> 5, lane = threadIdx.x & 31;
  int row  = blockIdx.x * 8 + wave;
  const float* x = v + (size_t)row * E_;
  float xv[16]; float s = 0.f, s2 = 0.f;
#pragma unroll
  for (int t = 0; t < 16; ++t) { float a = x[t*32 + lane]; xv[t] = a; s += a; s2 += a*a; }
#pragma unroll
  for (int m = 16; m; m >>= 1) { s += __shfl_xor(s, m, 32); s2 += __shfl_xor(s2, m, 32); }
  float mean = s * (1.f/E_);
  float var  = s2 * (1.f/E_) - mean*mean;
  float rstd = rsqrtf(var + 1e-5f);
  h16* o = out + (size_t)row * E_;
#pragma unroll
  for (int t = 0; t < 16; ++t) {
    int e = t*32 + lane;
    o[e] = (h16)((xv[t]-mean)*rstd*gamma[e] + beta[e]);
  }
}

// ------------------- qrpe[bh,n,j] = q_head_row . rpe_emb[j] ----------------
__global__ __launch_bounds__(256) void qrpe_kernel(
    const float* __restrict__ q, const float* __restrict__ rpe,
    float* __restrict__ qrpe) {
  __shared__ float qs[8][64];
  int wave = threadIdx.x >> 5, lane = threadIdx.x & 31;
  int row = blockIdx.x * 8 + wave;          // row = bh*N + n
  int bh = row / N_, n = row % N_;
  int b = bh >> 3, h = bh & 7;
  const float* qrow = q + ((size_t)(b*N_ + n))*E_ + h*D_;
  qs[wave][lane]    = qrow[lane];
  qs[wave][lane+32] = qrow[lane+32];
  __syncthreads();
#pragma unroll
  for (int t = 0; t < 5; ++t) {
    int j = lane + t*32;
    if (j < RPE_N) {
      float acc = 0.f;
#pragma unroll
      for (int dd = 0; dd < 64; ++dd) acc += rpe[j*64 + dd] * qs[wave][dd];
      qrpe[(size_t)row*RPE_STRIDE + j] = acc;
    }
  }
}

// ----------------------------- flash attention -----------------------------
__global__ __launch_bounds__(256) void flash_attn_kernel(
    const h16* __restrict__ q16, const h16* __restrict__ k16,
    const h16* __restrict__ v16, const float* __restrict__ qrpe,
    h16* __restrict__ attn_out) {
  __shared__ __align__(16) h16 Qs[128][72];   // queries   (row-major, +8 pad)
  __shared__ __align__(16) h16 Ks[64][72];    // keys      (row-major)
  __shared__ __align__(16) h16 Vt[64][72];    // values    (feature-major)
  __shared__ __align__(16) h16 Ps[8][16][72]; // per-wave probability tile

  const int tid  = threadIdx.x;
  const int wave = tid >> 5, lane = tid & 31;
  const int hl   = lane >> 4;                 // half-wave (C/D row offset +8)
  const int lm   = lane & 15;
  const int qt = blockIdx.x, bh = blockIdx.y;
  const int b = bh >> 3, h = bh & 7;
  const int q0 = qt * 128;

  { // stage Q tile 128x64 via async global->LDS (16B segments)
    const h16* qbase = q16 + ((size_t)(b*N_ + q0))*E_ + h*D_;
    for (int idx = tid; idx < 128*8; idx += 256) {
      int row = idx >> 3, seg = idx & 7;
      async_copy16(&Qs[row][seg*8], qbase + (size_t)row*E_ + seg*8);
    }
  }
  async_wait();
  __syncthreads();

  // per-wave Q A-fragments: halves0-7 <- K=k0..k0+7, halves8-15 <- K=k0+16..k0+23
  v16h aq[2];
  {
    int m = wave*16 + lm;
#pragma unroll
    for (int c = 0; c < 2; ++c) {
      int k0 = hl*8 + c*32;
      aq[c] = join16(*(const v8h*)&Qs[m][k0], *(const v8h*)&Qs[m][k0+16]);
    }
  }

  v8f o[4] = {};
  float mstat[8], lstat[8];
#pragma unroll
  for (int r = 0; r < 8; ++r) { mstat[r] = -1e30f; lstat[r] = 0.f; }

  // hoisted per-row RPE base pointers (kills per-element 64-bit addr math)
  int mgr[8]; const float* rrow[8];
#pragma unroll
  for (int r = 0; r < 8; ++r) {
    mgr[r]  = q0 + wave*16 + r + hl*8;
    rrow[r] = qrpe + ((size_t)bh*N_ + mgr[r])*RPE_STRIDE + 64;
  }

  const h16* kbase0 = k16 + ((size_t)(b*N_))*E_ + h*D_;
  const h16* vbase0 = v16 + ((size_t)(b*N_))*E_ + h*D_;

  for (int kt = 0; kt < N_/64; ++kt) {
    __syncthreads();
    { // stage K (async, row-major) and V (scalar transpose) tiles
      const h16* kbase = kbase0 + (size_t)(kt*64)*E_;
      for (int idx = tid; idx < 64*8; idx += 256) {
        int row = idx >> 3, seg = idx & 7;
        async_copy16(&Ks[row][seg*8], kbase + (size_t)row*E_ + seg*8);
      }
      const h16* vbase = vbase0 + (size_t)(kt*64)*E_;
      for (int idx = tid; idx < 64*64; idx += 256) {
        int key = idx >> 6, feat = idx & 63;
        Vt[feat][key] = vbase[(size_t)key*E_ + feat];
      }
    }
    async_wait();
    __syncthreads();

    // prefetch next tile's K/V rows into L2/L0 (one 128B row per thread)
    if (kt + 1 < N_/64) {
      const h16* nk = kbase0 + (size_t)((kt+1)*64)*E_;
      const h16* nv = vbase0 + (size_t)((kt+1)*64)*E_;
      __builtin_prefetch(nk + (size_t)(tid & 63)*E_, 0, 3);
      __builtin_prefetch(nv + (size_t)(tid & 63)*E_, 0, 3);
    }

    // S = Q K^T : 4 tiles of 16x16 across 64 keys, K=64 in 2 chunks
    v8f s[4];
#pragma unroll
    for (int tc = 0; tc < 4; ++tc) {
      v8f acc = {};
      int key = tc*16 + lm;
#pragma unroll
      for (int c = 0; c < 2; ++c) {
        int k0 = hl*8 + c*32;    // B frag: K=k0..k0+7, k0+16..k0+23 per lane
        v16h bk = join16(*(const v8h*)&Ks[key][k0], *(const v8h*)&Ks[key][k0+16]);
        acc = __builtin_amdgcn_wmma_f32_16x16x32_f16(false, aq[c], false, bk,
                                                     (short)0, acc, false, false);
      }
      s[tc] = acc;
    }

    // add RPE bias (gather), apply 1/sqrt(d)
#pragma unroll
    for (int tc = 0; tc < 4; ++tc) {
      int key = kt*64 + tc*16 + lm;
#pragma unroll
      for (int r = 0; r < 8; ++r) {
        int di = key - mgr[r]; di = di < -64 ? -64 : (di > 64 ? 64 : di);
        s[tc][r] = (s[tc][r] + rrow[r][di]) * 0.125f;
      }
    }

    // current row max (reduce within each 16-lane half: rows r / r+8)
    float rmax[8];
#pragma unroll
    for (int r = 0; r < 8; ++r) {
      float mx = fmaxf(fmaxf(s[0][r], s[1][r]), fmaxf(s[2][r], s[3][r]));
#pragma unroll
      for (int msk = 8; msk; msk >>= 1) mx = fmaxf(mx, __shfl_xor(mx, msk, 32));
      rmax[r] = mx;
    }

    // online-softmax rescale of running state + accumulators
#pragma unroll
    for (int r = 0; r < 8; ++r) {
      float mnew  = fmaxf(mstat[r], rmax[r]);
      float alpha = __expf(mstat[r] - mnew);
      mstat[r] = mnew;
      lstat[r] *= alpha;
#pragma unroll
      for (int ft = 0; ft < 4; ++ft) o[ft][r] *= alpha;
    }

    // P = exp(S - m); spill f16 tile to LDS for A-fragment re-layout
#pragma unroll
    for (int tc = 0; tc < 4; ++tc) {
#pragma unroll
      for (int r = 0; r < 8; ++r) {
        float p = __expf(s[tc][r] - mstat[r]);
        s[tc][r] = p;
        Ps[wave][r + hl*8][tc*16 + lm] = (h16)p;
      }
    }
#pragma unroll
    for (int r = 0; r < 8; ++r) {
      float sm = s[0][r] + s[1][r] + s[2][r] + s[3][r];
#pragma unroll
      for (int msk = 8; msk; msk >>= 1) sm += __shfl_xor(sm, msk, 32);
      lstat[r] += sm;
    }
    __syncthreads();   // order Ps stores before A-fragment reads

    // O += P @ V
    v16h pa[2];
#pragma unroll
    for (int c = 0; c < 2; ++c) {
      int k0 = hl*8 + c*32;
      pa[c] = join16(*(const v8h*)&Ps[wave][lm][k0], *(const v8h*)&Ps[wave][lm][k0+16]);
    }
#pragma unroll
    for (int ft = 0; ft < 4; ++ft) {
      int feat = ft*16 + lm;
#pragma unroll
      for (int c = 0; c < 2; ++c) {
        int k0 = hl*8 + c*32;
        v16h bv = join16(*(const v8h*)&Vt[feat][k0], *(const v8h*)&Vt[feat][k0+16]);
        o[ft] = __builtin_amdgcn_wmma_f32_16x16x32_f16(false, pa[c], false, bv,
                                                       (short)0, o[ft], false, false);
      }
    }
  }

  // normalize by row sums; store f16 in [b, n, (h d)] layout for projection
#pragma unroll
  for (int r = 0; r < 8; ++r) lstat[r] = 1.f / lstat[r];
  h16* obase = attn_out + ((size_t)(b*N_))*E_ + h*D_;
#pragma unroll
  for (int ft = 0; ft < 4; ++ft) {
#pragma unroll
    for (int r = 0; r < 8; ++r) {
      obase[(size_t)mgr[r]*E_ + ft*16 + lm] = (h16)(o[ft][r] * lstat[r]);
    }
  }
}

// --------------------- output projection: out = A @ W^T --------------------
__global__ __launch_bounds__(256) void proj_kernel(
    const h16* __restrict__ a16, const h16* __restrict__ w16,
    float* __restrict__ out) {
  __shared__ __align__(16) h16 As[128][40];
  __shared__ __align__(16) h16 Bs[128][40];
  const int tid = threadIdx.x, wave = tid >> 5, lane = tid & 31;
  const int hl = lane >> 4, lm = lane & 15;
  const int i0 = blockIdx.x * 128, j0 = blockIdx.y * 128;

  v8f c[8] = {};
  for (int kc = 0; kc < 16; ++kc) {
    __syncthreads();
    for (int idx = tid; idx < 128*4; idx += 256) {
      int row = idx >> 2, seg = idx & 3;
      async_copy16(&As[row][seg*8], a16 + (size_t)(i0+row)*E_ + kc*32 + seg*8);
      async_copy16(&Bs[row][seg*8], w16 + (size_t)(j0+row)*E_ + kc*32 + seg*8);
    }
    async_wait();
    __syncthreads();
    v16h a;
    {
      int row = wave*16 + lm, k0 = hl*8;
      a = join16(*(const v8h*)&As[row][k0], *(const v8h*)&As[row][k0+16]);
    }
#pragma unroll
    for (int jt = 0; jt < 8; ++jt) {
      int col = jt*16 + lm, k0 = hl*8;
      v16h bb = join16(*(const v8h*)&Bs[col][k0], *(const v8h*)&Bs[col][k0+16]);
      c[jt] = __builtin_amdgcn_wmma_f32_16x16x32_f16(false, a, false, bb,
                                                     (short)0, c[jt], false, false);
    }
  }
#pragma unroll
  for (int jt = 0; jt < 8; ++jt)
#pragma unroll
    for (int r = 0; r < 8; ++r)
      out[(size_t)(i0 + wave*16 + r + hl*8)*E_ + j0 + jt*16 + lm] = c[jt][r];
}

// ------------------------------- launcher ----------------------------------
extern "C" void kernel_launch(void* const* d_in, const int* in_sizes, int n_in,
                              void* d_out, int out_size, void* d_ws, size_t ws_size,
                              hipStream_t stream) {
  const float* q   = (const float*)d_in[0];
  const float* k   = (const float*)d_in[1];
  const float* v   = (const float*)d_in[2];
  const float* g   = (const float*)d_in[3];
  const float* bta = (const float*)d_in[4];
  const float* rpe = (const float*)d_in[5];
  const float* W   = (const float*)d_in[6];
  // d_in[7] = rel_idx: recomputed analytically on device (clip(m-n)+64)

  char* ws = (char*)d_ws;
  size_t off = 0;
  auto carve = [&](size_t bytes) {
    void* p = ws + off; off += (bytes + 255) & ~(size_t)255; return p;
  };
  h16*   q16 = (h16*)  carve((size_t)B_*N_*E_*2);
  h16*   k16 = (h16*)  carve((size_t)B_*N_*E_*2);
  h16*   v16 = (h16*)  carve((size_t)B_*N_*E_*2);
  h16*   w16 = (h16*)  carve((size_t)E_*E_*2);
  float* qrp = (float*)carve((size_t)BH_*N_*RPE_STRIDE*4);
  h16*   a16 = (h16*)  carve((size_t)B_*N_*E_*2);

  const int cnt = B_*N_*E_;
  f32_to_f16_kernel<<<cnt/256, 256, 0, stream>>>(q, q16, cnt);
  f32_to_f16_kernel<<<cnt/256, 256, 0, stream>>>(k, k16, cnt);
  f32_to_f16_kernel<<<(E_*E_)/256, 256, 0, stream>>>(W, w16, E_*E_);
  layernorm_v_kernel<<<(B_*N_)/8, 256, 0, stream>>>(v, g, bta, v16);
  qrpe_kernel<<<(BH_*N_)/8, 256, 0, stream>>>(q, rpe, qrp);
  flash_attn_kernel<<<dim3(N_/128, BH_), 256, 0, stream>>>(q16, k16, v16, qrp, a16);
  proj_kernel<<<dim3((B_*N_)/128, E_/128), 256, 0, stream>>>(a16, w16, (float*)d_out);
}
